// InterpolationStage_29059748725686
// MI455X (gfx1250) — compile-verified, matrix operand order checked by reading, not run on hardware
//
#include <hip/hip_runtime.h>

typedef __attribute__((ext_vector_type(16))) _Float16 v16h;
typedef __attribute__((ext_vector_type(8)))  _Float16 v8h;
typedef __attribute__((ext_vector_type(8)))  float    v8f;
typedef __attribute__((ext_vector_type(4)))  float    v4f;

#define NC   8192
#define NF   32768
#define DD   256
#define DE   128
#define DO   128

// ---------------------------------------------------------------------------
// One-shot f32 -> f16 conversion (8 elements/thread, vectorized).
// ---------------------------------------------------------------------------
__global__ __launch_bounds__(256) void cvt_f16_kernel(
    const float* __restrict__ src, _Float16* __restrict__ dst, int n8)
{
  const int i = blockIdx.x * 256 + threadIdx.x;
  if (i < n8) {
    v4f a = *(const v4f*)(src + (size_t)i * 8);
    v4f b = *(const v4f*)(src + (size_t)i * 8 + 4);
    v8h r;
#pragma unroll
    for (int t = 0; t < 4; ++t) { r[t] = (_Float16)a[t]; r[4 + t] = (_Float16)b[t]; }
    *(v8h*)(dst + (size_t)i * 8) = r;
  }
}

// ---------------------------------------------------------------------------
// Fragment loader from pre-converted f16 rows: two aligned 16B chunks.
// Lane group g: halves 0..7 = k0+8g.., halves 8..15 = k0+16+8g..
// ---------------------------------------------------------------------------
__device__ __forceinline__ v16h load_frag_h(const _Float16* __restrict__ row, int g) {
  v8h lo = *(const v8h*)(row + 8 * g);
  v8h hi = *(const v8h*)(row + 16 + 8 * g);
  v16h r;
#pragma unroll
  for (int t = 0; t < 8; ++t) { r[t] = lo[t]; r[8 + t] = hi[t]; }
  return r;
}

// ---------------------------------------------------------------------------
// C[M,256] = A[M,Kd] @ W[256,Kd]^T + bias (A, W pre-converted f16).
// One wave -> 16x64 tile; 4 waves/block.
// ---------------------------------------------------------------------------
__global__ __launch_bounds__(128) void gemm_bias_kernel(
    const _Float16* __restrict__ A, const _Float16* __restrict__ W,
    const float* __restrict__ bias, float* __restrict__ C, int Kd)
{
  const int lane = threadIdx.x & 31;
  const int wid  = threadIdx.x >> 5;
  const int tile = blockIdx.x * 4 + wid;
  const int mbase = (tile >> 2) * 16;
  const int nbase = (tile & 3) * 64;
  const int g = lane >> 4, j = lane & 15;

  v8f acc[4] = {};
  const _Float16* arow = A + (size_t)(mbase + j) * Kd;
  for (int k0 = 0; k0 < Kd; k0 += 32) {
    v16h af = load_frag_h(arow + k0, g);
#pragma unroll
    for (int t = 0; t < 4; ++t) {
      const _Float16* brow = W + (size_t)(nbase + 16 * t + j) * Kd + k0;
      v16h bf = load_frag_h(brow, g);
      acc[t] = __builtin_amdgcn_wmma_f32_16x16x32_f16(
          false, af, false, bf, (short)0, acc[t], false, false);
    }
  }
#pragma unroll
  for (int t = 0; t < 4; ++t) {
    float bn = bias[nbase + 16 * t + j];
    float* cp = C + (size_t)mbase * 256 + nbase + 16 * t + j;
#pragma unroll
    for (int e = 0; e < 8; ++e) cp[(size_t)(e + 8 * g) * 256] = acc[t][e] + bn;
  }
}

// ---------------------------------------------------------------------------
// Brute-force KNN: 16 nearest coarse points per fine point.
// Coarse positions (96 KB) staged in LDS; one wave per fine point (x8 iters).
// ---------------------------------------------------------------------------
__global__ __launch_bounds__(256) void knn_kernel(
    const float* __restrict__ dpos, const float* __restrict__ epos,
    int* __restrict__ knn_idx)
{
  __shared__ float px[NC], py[NC], pz[NC];
  for (int t = threadIdx.x; t < NC; t += 256) {
    px[t] = dpos[3 * t]; py[t] = dpos[3 * t + 1]; pz[t] = dpos[3 * t + 2];
  }
  __syncthreads();

  const int lane = threadIdx.x & 31;
  const int wid  = threadIdx.x >> 5;
  for (int it = 0; it < 8; ++it) {
    const int n = blockIdx.x * 64 + wid * 8 + it;
    const float qx = epos[3 * n], qy = epos[3 * n + 1], qz = epos[3 * n + 2];

    float bd[16]; int bi[16];
#pragma unroll
    for (int t = 0; t < 16; ++t) { bd[t] = 3.0e38f; bi[t] = 0; }
    float worst = 3.0e38f;

    for (int jj = 0; jj < NC / 32; ++jj) {
      const int c = (jj << 5) + lane;
      const float dx = px[c] - qx, dy = py[c] - qy, dz = pz[c] - qz;
      const float d = dx * dx + dy * dy + dz * dz;
      if (d < worst) {
        bool done = false;
#pragma unroll
        for (int t = 0; t < 16; ++t)
          if (!done && bd[t] == worst) { bd[t] = d; bi[t] = c; done = true; }
        worst = bd[0];
#pragma unroll
        for (int t = 1; t < 16; ++t) worst = fmaxf(worst, bd[t]);
      }
    }

    int my_out = 0;
    for (int r = 0; r < 16; ++r) {
      float lmin = 3.0e38f; int lix = 0;
#pragma unroll
      for (int t = 0; t < 16; ++t)
        if (bd[t] < lmin) { lmin = bd[t]; lix = bi[t]; }
      float v = lmin; int wl = lane; int wi = lix;
#pragma unroll
      for (int mask = 16; mask >= 1; mask >>= 1) {
        float ov = __shfl_xor(v, mask, 32);
        int owl = __shfl_xor(wl, mask, 32);
        int owi = __shfl_xor(wi, mask, 32);
        if (ov < v || (ov == v && owl < wl)) { v = ov; wl = owl; wi = owi; }
      }
      if (lane == wl) {
        bool done = false;
#pragma unroll
        for (int t = 0; t < 16; ++t)
          if (!done && bd[t] == v) { bd[t] = 3.0e38f; done = true; }
      }
      if (lane == r) my_out = wi;
    }
    if (lane < 16) knn_idx[n * 16 + lane] = my_out;
  }
}

// ---------------------------------------------------------------------------
// Attention: one wave per fine point; gathers from L2-resident K_all/V_all.
// Emits agg directly as f16 (WMMA A-operand of the MLP).
// ---------------------------------------------------------------------------
__global__ __launch_bounds__(256) void attn_kernel(
    const float* __restrict__ query, const float* __restrict__ Kall,
    const float* __restrict__ Vall, const int* __restrict__ knn_idx,
    _Float16* __restrict__ agg_h)
{
  const int lane = threadIdx.x & 31;
  const int wid  = threadIdx.x >> 5;
  const int n = blockIdx.x * 8 + wid;

  float q8[8];
  const float* q = query + (size_t)n * 256 + lane * 8;
#pragma unroll
  for (int t = 0; t < 8; ++t) q8[t] = q[t];

  int ids[16]; float s[16];
#pragma unroll
  for (int e = 0; e < 16; ++e) ids[e] = knn_idx[n * 16 + e];

#pragma unroll
  for (int e = 0; e < 16; ++e) {
    const float* kr = Kall + (size_t)ids[e] * 256 + lane * 8;
    float p = 0.f;
#pragma unroll
    for (int t = 0; t < 8; ++t) p += q8[t] * kr[t];
#pragma unroll
    for (int mask = 16; mask >= 1; mask >>= 1) p += __shfl_xor(p, mask, 32);
    s[e] = p * 0.0625f;  // 1/sqrt(256)
  }
  float mx = s[0];
#pragma unroll
  for (int e = 1; e < 16; ++e) mx = fmaxf(mx, s[e]);
  float sum = 0.f;
#pragma unroll
  for (int e = 0; e < 16; ++e) { s[e] = __expf(s[e] - mx); sum += s[e]; }
  const float inv = 1.f / sum;

  float a8[8] = {};
#pragma unroll
  for (int e = 0; e < 16; ++e) {
    const float w = s[e] * inv;
    const float* vr = Vall + (size_t)ids[e] * 256 + lane * 8;
#pragma unroll
    for (int t = 0; t < 8; ++t) a8[t] += w * vr[t];
  }
  v8h r;
#pragma unroll
  for (int t = 0; t < 8; ++t) r[t] = (_Float16)a8[t];
  *(v8h*)(agg_h + (size_t)n * 256 + lane * 8) = r;
}

// ---------------------------------------------------------------------------
// Fused MLP + LayerNorm. One wave -> 16 rows x 128 out.
//   h  = relu([agg|ef] @ W1^T + b1)   (K=384, WMMA, h -> LDS f16)
//   up = h @ W2^T + b2                (K=128, WMMA, A-frags via ds_load_b128)
//   out = LN(up) * g + b              (shfl over 16-lane groups)
// ---------------------------------------------------------------------------
__global__ __launch_bounds__(128) void mlp_ln_kernel(
    const _Float16* __restrict__ agg_h, const _Float16* __restrict__ ef_h,
    const _Float16* __restrict__ W1h, const float* __restrict__ b1,
    const _Float16* __restrict__ W2h, const float* __restrict__ b2,
    const float* __restrict__ ln_g, const float* __restrict__ ln_b,
    float* __restrict__ out)
{
  __shared__ _Float16 hb[4][16][136];   // 272B rows = 17x16B: aligned + rotating banks
  const int lane = threadIdx.x & 31;
  const int wid  = threadIdx.x >> 5;
  const int mbase = (blockIdx.x * 4 + wid) * 16;
  const int g = lane >> 4, j = lane & 15;

  v8f acc[8] = {};
  for (int ks = 0; ks < 12; ++ks) {
    const int k0 = ks * 32;
    const _Float16* arow = (k0 < 256)
        ? agg_h + (size_t)(mbase + j) * 256 + k0
        : ef_h  + (size_t)(mbase + j) * 128 + (k0 - 256);
    v16h af = load_frag_h(arow, g);
#pragma unroll
    for (int t = 0; t < 8; ++t) {
      const _Float16* brow = W1h + (size_t)(16 * t + j) * 384 + k0;
      v16h bf = load_frag_h(brow, g);
      acc[t] = __builtin_amdgcn_wmma_f32_16x16x32_f16(
          false, af, false, bf, (short)0, acc[t], false, false);
    }
  }
#pragma unroll
  for (int t = 0; t < 8; ++t) {
    const float bn = b1[16 * t + j];
#pragma unroll
    for (int e = 0; e < 8; ++e) {
      float v = acc[t][e] + bn;
      hb[wid][e + 8 * g][16 * t + j] = (_Float16)(v > 0.f ? v : 0.f);
    }
  }
  __syncthreads();

  v8f acc2[8] = {};
#pragma unroll
  for (int ks = 0; ks < 4; ++ks) {
    const int k0 = ks * 32;
    v16h af = load_frag_h(&hb[wid][j][k0], g);
#pragma unroll
    for (int t = 0; t < 8; ++t) {
      const _Float16* brow = W2h + (size_t)(16 * t + j) * 128 + k0;
      v16h bf = load_frag_h(brow, g);
      acc2[t] = __builtin_amdgcn_wmma_f32_16x16x32_f16(
          false, af, false, bf, (short)0, acc2[t], false, false);
    }
  }

  float su[8] = {}, sq[8] = {};
#pragma unroll
  for (int t = 0; t < 8; ++t) {
    const float bn = b2[16 * t + j];
#pragma unroll
    for (int e = 0; e < 8; ++e) {
      const float v = acc2[t][e] + bn;
      acc2[t][e] = v;
      su[e] += v; sq[e] += v * v;
    }
  }
#pragma unroll
  for (int mask = 8; mask >= 1; mask >>= 1) {
#pragma unroll
    for (int e = 0; e < 8; ++e) {
      su[e] += __shfl_xor(su[e], mask, 32);
      sq[e] += __shfl_xor(sq[e], mask, 32);
    }
  }
#pragma unroll
  for (int t = 0; t < 8; ++t) {
    const float gam = ln_g[16 * t + j], bet = ln_b[16 * t + j];
#pragma unroll
    for (int e = 0; e < 8; ++e) {
      const float mean = su[e] * (1.f / 128.f);
      const float var  = sq[e] * (1.f / 128.f) - mean * mean;
      const float v = (acc2[t][e] - mean) * rsqrtf(var + 1e-5f) * gam + bet;
      out[(size_t)(mbase + e + 8 * g) * 128 + 16 * t + j] = v;
    }
  }
}

// ---------------------------------------------------------------------------
extern "C" void kernel_launch(void* const* d_in, const int* in_sizes, int n_in,
                              void* d_out, int out_size, void* d_ws, size_t ws_size,
                              hipStream_t stream) {
  (void)in_sizes; (void)n_in; (void)ws_size;
  const float* ddf  = (const float*)d_in[0];
  const float* dpos = (const float*)d_in[1];
  const float* ef   = (const float*)d_in[2];
  const float* epos = (const float*)d_in[3];
  const void*  labels = d_in[4];
  const float* Wq = (const float*)d_in[5];
  const float* bq = (const float*)d_in[6];
  const float* Wk = (const float*)d_in[7];
  const float* bk = (const float*)d_in[8];
  const float* Wv = (const float*)d_in[9];
  const float* bv = (const float*)d_in[10];
  const float* W1 = (const float*)d_in[11];
  const float* b1 = (const float*)d_in[12];
  const float* W2 = (const float*)d_in[13];
  const float* b2 = (const float*)d_in[14];
  const float* lg = (const float*)d_in[15];
  const float* lb = (const float*)d_in[16];

  char* ws = (char*)d_ws;
  float*    Kall  = (float*)   (ws);                        //  8 MB
  float*    Vall  = (float*)   (ws + ((size_t)8  << 20));   //  8 MB
  float*    query = (float*)   (ws + ((size_t)16 << 20));   // 32 MB
  _Float16* agg_h = (_Float16*)(ws + ((size_t)48 << 20));   // 16 MB
  int*      kidx  = (int*)     (ws + ((size_t)64 << 20));   //  2 MB
  _Float16* df_h  = (_Float16*)(ws + ((size_t)66 << 20));   //  4 MB
  _Float16* ef_h  = (_Float16*)(ws + ((size_t)70 << 20));   //  8 MB
  _Float16* Wk_h  = (_Float16*)(ws + ((size_t)78 << 20));   // 128 KB
  _Float16* Wv_h  = Wk_h + (size_t)DD * DD;                 // 128 KB
  _Float16* Wq_h  = Wv_h + (size_t)DD * DD;                 //  64 KB
  _Float16* W1_h  = Wq_h + (size_t)DD * DE;                 //  96 KB
  _Float16* W2_h  = W1_h + (size_t)DO * (DD + DE);          //  32 KB

  // One-shot f32->f16 pre-pass (features + weights).
  auto cvt = [&](const float* s, _Float16* d, size_t n) {
    int n8 = (int)(n / 8);
    cvt_f16_kernel<<<(n8 + 255) / 256, 256, 0, stream>>>(s, d, n8);
  };
  cvt(ddf, df_h, (size_t)NC * DD);
  cvt(ef,  ef_h, (size_t)NF * DE);
  cvt(Wk,  Wk_h, (size_t)DD * DD);
  cvt(Wv,  Wv_h, (size_t)DD * DD);
  cvt(Wq,  Wq_h, (size_t)DD * DE);
  cvt(W1,  W1_h, (size_t)DO * (DD + DE));
  cvt(W2,  W2_h, (size_t)DO * DO);

  knn_kernel<<<512, 256, 0, stream>>>(dpos, epos, kidx);
  gemm_bias_kernel<<<512, 128, 0, stream>>>(df_h, Wk_h, bk, Kall, DD);
  gemm_bias_kernel<<<512, 128, 0, stream>>>(df_h, Wv_h, bv, Vall, DD);
  gemm_bias_kernel<<<2048, 128, 0, stream>>>(ef_h, Wq_h, bq, query, DE);
  attn_kernel<<<4096, 256, 0, stream>>>(query, Kall, Vall, kidx, agg_h);
  mlp_ln_kernel<<<512, 128, 0, stream>>>(agg_h, ef_h, W1_h, b1, W2_h, b2,
                                         lg, lb, (float*)d_out);

  // pass-through outputs: encoder_pos, encoder_labels
  float* out_pos = (float*)d_out + (size_t)NF * DO;
  hipMemcpyAsync(out_pos, epos, (size_t)NF * 3 * sizeof(float),
                 hipMemcpyDeviceToDevice, stream);
  long long rem = (long long)out_size - (long long)NF * DO - (long long)NF * 3;
  if (rem > 0) {
    hipMemcpyAsync((float*)d_out + (size_t)NF * (DO + 3), labels,
                   (size_t)rem * 4, hipMemcpyDeviceToDevice, stream);
  }
}
